// EncoderLayer_62895501082866
// MI455X (gfx1250) — matrix operand order repaired and checked by reading, not compiled, not run on hardware
//
#include <hip/hip_runtime.h>
#include <hip/hip_bf16.h>
#include <stdint.h>

// ---------------------------------------------------------------------------
// Types for CDNA5 WMMA (wave32)
// ---------------------------------------------------------------------------
typedef __attribute__((ext_vector_type(16))) __bf16 v16bf;
typedef __attribute__((ext_vector_type(8)))  float  v8f;

union FragBF {
    v16bf v;
    uint4 q[2];
};
static_assert(sizeof(v16bf) == 32, "v16bf must be 32 bytes");

__device__ __forceinline__ unsigned short f2bf(float f) {
    unsigned int u = __float_as_uint(f);
    u += 0x7FFFu + ((u >> 16) & 1u);   // round-to-nearest-even
    return (unsigned short)(u >> 16);
}

__device__ __forceinline__ v8f wmma_bf16(const FragBF& a, const FragBF& b, v8f c) {
    return __builtin_amdgcn_wmma_f32_16x16x32_bf16(
        /*neg_a=*/false, a.v, /*neg_b=*/false, b.v,
        /*c_mod=*/(short)0, c, /*reuse_a=*/false, /*reuse_b=*/false);
}

// ---------------------------------------------------------------------------
// CDNA5 async global->LDS copy (16B per lane), tracked with ASYNCcnt.
// LDS byte address = low 32 bits of the flat address (LDS aperture).
// ---------------------------------------------------------------------------
__device__ __forceinline__ void async_ld16(void* lds, const void* g) {
    asm volatile("global_load_async_to_lds_b128 %0, %1, off"
                 :: "v"((unsigned)(uintptr_t)lds),
                    "v"((unsigned long long)(uintptr_t)g)
                 : "memory");
}
__device__ __forceinline__ void wait_async0() {
    asm volatile("s_wait_asynccnt 0x0" ::: "memory");
}
__device__ __forceinline__ void wait_async2() {
    asm volatile("s_wait_asynccnt 0x2" ::: "memory");
}
__device__ __forceinline__ void wait_async4() {
    asm volatile("s_wait_asynccnt 0x4" ::: "memory");
}

// ---------------------------------------------------------------------------
// Problem constants
// ---------------------------------------------------------------------------
#define BB 2
#define LL 2048
#define DD 1024
#define FF 4096
#define HH 16
#define DKH 64
#define MM (BB*LL)   // 4096 rows

// ---------------------------------------------------------------------------
// fp32 -> bf16 convert (x)
// ---------------------------------------------------------------------------
__global__ __launch_bounds__(256)
void cvtx_kernel(const float* __restrict__ x, unsigned short* __restrict__ xb) {
    int i = blockIdx.x * blockDim.x + threadIdx.x;          // over M*D/4
    float4 f = ((const float4*)x)[i];
    uint2 o;
    o.x = (unsigned int)f2bf(f.x) | ((unsigned int)f2bf(f.y) << 16);
    o.y = (unsigned int)f2bf(f.z) | ((unsigned int)f2bf(f.w) << 16);
    ((uint2*)xb)[i] = o;
}

// ---------------------------------------------------------------------------
// Transpose+convert weight: W[K,N] fp32 -> Wt[N,K] bf16   (tiled via LDS)
// ---------------------------------------------------------------------------
__global__ __launch_bounds__(256)
void wtrans_kernel(const float* __restrict__ W, unsigned short* __restrict__ Wt,
                   int K, int N) {
    __shared__ float tile[32][33];
    const int k0 = blockIdx.x * 32;
    const int n0 = blockIdx.y * 32;
    const int tx = threadIdx.x;      // 0..31
    const int ty = threadIdx.y;      // 0..7
#pragma unroll
    for (int i = 0; i < 4; i++)
        tile[ty + 8*i][tx] = W[(size_t)(k0 + ty + 8*i) * N + n0 + tx];
    __syncthreads();
#pragma unroll
    for (int i = 0; i < 4; i++)
        Wt[(size_t)(n0 + ty + 8*i) * K + k0 + tx] = f2bf(tile[tx][ty + 8*i]);
}

// ---------------------------------------------------------------------------
// Generic bf16 WMMA GEMM:  out[M,N] = A[M,K] * Bt[N,K]^T (+bias)(+res)(relu)
// Block tile 128x128, BK=32, 256 threads = 8 waves, wave tile 32x64.
// Tiles staged with async global->LDS (double buffered, ASYNCcnt pipelined).
// ---------------------------------------------------------------------------
template<bool BIAS, bool RES, bool RELU, bool BF16OUT>
__global__ __launch_bounds__(256)
void gemm_bf16(const unsigned short* __restrict__ A,
               const unsigned short* __restrict__ Bt,
               const float* __restrict__ bias,
               const float* __restrict__ res,
               float* __restrict__ outF,
               unsigned short* __restrict__ outB,
               int M, int N, int K) {
    constexpr int BM = 128, BN = 128, BK = 32, LDSS = 40;   // 8-elem pad
    __shared__ __align__(16) unsigned short lA[2][BM * LDSS];
    __shared__ __align__(16) unsigned short lB[2][BN * LDSS];

    const int nblk = N / BN;
    const int m0 = (blockIdx.x / nblk) * BM;
    const int n0 = (blockIdx.x % nblk) * BN;

    const int tid  = threadIdx.x;
    const int wid  = tid >> 5;
    const int lane = tid & 31;
    const int ln16 = lane & 15;
    const int sel  = lane >> 4;
    const int waveM = wid >> 1;      // 0..3 -> 32-row strip
    const int waveN = wid & 1;       // 0..1 -> 64-col strip

    v8f acc[2][4];
    const v8f vzero = {};
#pragma unroll
    for (int i = 0; i < 2; i++)
#pragma unroll
        for (int j = 0; j < 4; j++) acc[i][j] = vzero;

    const int ldRow  = tid >> 1;             // 0..127
    const int ldHalf = (tid & 1) * 16;       // 0 or 16 elements
    const unsigned short* aG = A  + (size_t)(m0 + ldRow) * K + ldHalf;
    const unsigned short* bG = Bt + (size_t)(n0 + ldRow) * K + ldHalf;
    const int ldsOff = ldRow * LDSS + ldHalf;

    // issue one 32-wide K slab (4 x b128 per thread) into buffer `buf`
    auto issue = [&](int k0, int buf) {
        async_ld16(&lA[buf][ldsOff],     aG + k0);
        async_ld16(&lA[buf][ldsOff + 8], aG + k0 + 8);
        async_ld16(&lB[buf][ldsOff],     bG + k0);
        async_ld16(&lB[buf][ldsOff + 8], bG + k0 + 8);
    };

    const int ntiles = K / BK;
    issue(0, 0);
    for (int t = 0; t < ntiles; ++t) {
        const int buf = t & 1;
        if (t + 1 < ntiles) {
            issue((t + 1) * BK, buf ^ 1);
            wait_async4();          // in-order: previous slab's 4 ops complete
        } else {
            wait_async0();
        }
        __syncthreads();            // slab visible to all waves

        FragBF afr[2], bfr[4];
#pragma unroll
        for (int mt = 0; mt < 2; mt++) {
            int base = (waveM*32 + mt*16 + ln16) * LDSS + sel*8;
            afr[mt].q[0] = *(const uint4*)(&lA[buf][base]);        // K sel*8+0..7
            afr[mt].q[1] = *(const uint4*)(&lA[buf][base + 16]);   // K 16+sel*8..
        }
#pragma unroll
        for (int nt = 0; nt < 4; nt++) {
            int base = (waveN*64 + nt*16 + ln16) * LDSS + sel*8;
            bfr[nt].q[0] = *(const uint4*)(&lB[buf][base]);        // K sel*8+0..7
            bfr[nt].q[1] = *(const uint4*)(&lB[buf][base + 16]);   // K 16+sel*8..
        }
#pragma unroll
        for (int mt = 0; mt < 2; mt++)
#pragma unroll
            for (int nt = 0; nt < 4; nt++)
                acc[mt][nt] = wmma_bf16(afr[mt], bfr[nt], acc[mt][nt]);

        __syncthreads();            // all waves done with buf before overwrite
    }

    // ---- epilogue: C layout = row (lane/16)*8+v (VGPR v), col lane%16 ----
#pragma unroll
    for (int mt = 0; mt < 2; mt++) {
#pragma unroll
        for (int nt = 0; nt < 4; nt++) {
            const int col   = n0 + waveN*64 + nt*16 + ln16;
            const int rbase = m0 + waveM*32 + mt*16 + sel*8;
            float bv = 0.f;
            if (BIAS) bv = bias[col];
#pragma unroll
            for (int v = 0; v < 8; v++) {
                const size_t idx = (size_t)(rbase + v) * N + col;
                float val = acc[mt][nt][v];
                if (BIAS) val += bv;
                if (RES)  val += res[idx];
                if (RELU) val = fmaxf(val, 0.f);
                if (BF16OUT) outB[idx] = f2bf(val);
                else         outF[idx] = val;
            }
        }
    }
}

// ---------------------------------------------------------------------------
// RoPE: qlin/klin fp32 [B,L,D] -> qh/kh bf16 [B,H,L,64]; q scaled by 1/sqrt(dk)
// ---------------------------------------------------------------------------
__global__ __launch_bounds__(256)
void rope_kernel(const float* __restrict__ qlin, const float* __restrict__ klin,
                 unsigned short* __restrict__ qh, unsigned short* __restrict__ kh) {
    int idx = blockIdx.x * blockDim.x + threadIdx.x;   // 2^21 threads
    int j = idx & 31;
    int l = (idx >> 5) & (LL - 1);
    int h = (idx >> 16) & (HH - 1);
    int b = idx >> 20;
    float inv = __expf(-(float)j * 0.28782314f);       // ln(10000)/32
    float ang = (float)l * inv;
    float s, c;
    __sincosf(ang, &s, &c);
    size_t src = ((size_t)(b * LL + l)) * DD + h * DKH;
    float q1 = qlin[src + j], q2 = qlin[src + j + 32];
    float k1 = klin[src + j], k2 = klin[src + j + 32];
    size_t dst = ((size_t)((b * HH + h) * LL + l)) * DKH;
    const float sc = 0.125f;                           // 1/sqrt(64)
    qh[dst + j]      = f2bf((q1 * c - q2 * s) * sc);
    qh[dst + j + 32] = f2bf((q1 * s + q2 * c) * sc);
    kh[dst + j]      = f2bf(k1 * c - k2 * s);
    kh[dst + j + 32] = f2bf(k1 * s + k2 * c);
}

// ---------------------------------------------------------------------------
// V transpose: vlin fp32 [B,L,D] -> vt bf16 [B,H,64,L]
// ---------------------------------------------------------------------------
__global__ __launch_bounds__(256)
void vtrans_kernel(const float* __restrict__ vlin, unsigned short* __restrict__ vt) {
    int idx = blockIdx.x * blockDim.x + threadIdx.x;   // 2^22 threads
    int l = idx & (LL - 1);
    int j = (idx >> 11) & (DKH - 1);
    int h = (idx >> 17) & (HH - 1);
    int b = idx >> 21;
    vt[idx] = f2bf(vlin[((size_t)(b * LL + l)) * DD + h * DKH + j]);
}

// ---------------------------------------------------------------------------
// Flash attention: 1 workgroup = (b,h, 128 q rows); 1 wave = 16 q rows.
// K/V tiles staged via async global->LDS (double buffered, shared by 8 waves).
// ---------------------------------------------------------------------------
__global__ __launch_bounds__(256)
void attn_kernel(const unsigned short* __restrict__ qh,
                 const unsigned short* __restrict__ kh,
                 const unsigned short* __restrict__ vt,
                 unsigned short* __restrict__ ctx) {
    constexpr int KP = 72;   // K tile row pitch (elems): 144B, conflict-free
    constexpr int VP = 40;   // V tile row pitch (elems): 80B
    __shared__ __align__(16) unsigned short lK[2][32 * KP];
    __shared__ __align__(16) unsigned short lV[2][64 * VP];
    __shared__ __align__(16) unsigned short lp[8 * 16 * 40];   // per-wave P tile

    const int blk = blockIdx.x;
    const int qt = blk & 15;
    const int h  = (blk >> 4) & (HH - 1);
    const int b  = blk >> 8;
    const int bh = b * HH + h;

    const int tid = threadIdx.x, wid = tid >> 5, lane = tid & 31;
    const int ln16 = lane & 15, hf = lane >> 4;
    const int qr0 = qt * 128 + wid * 16;

    // Q fragments (A layout), 2 k-steps over dk=64
    FragBF qf[2];
#pragma unroll
    for (int s = 0; s < 2; s++) {
        const unsigned short* p =
            qh + ((size_t)bh * LL + qr0 + ln16) * DKH + s * 32 + hf * 8;
        qf[s].q[0] = *(const uint4*)p;
        qf[s].q[1] = *(const uint4*)(p + 16);
    }

    // async staging assignment: K tile 32 rows x 64 elems, V tile 64 rows x 32
    const int kRow = tid >> 3, kChunk = (tid & 7) * 8;   // 8 chunks/row
    const int vRow = tid >> 2, vChunk = (tid & 3) * 8;   // 4 chunks/row
    const unsigned short* kBase = kh + ((size_t)bh * LL + kRow) * DKH + kChunk;
    const unsigned short* vBase = vt + ((size_t)bh * DKH + vRow) * LL + vChunk;
    auto issue = [&](int kt, int buf) {
        async_ld16(&lK[buf][kRow * KP + kChunk], kBase + (size_t)kt * DKH);
        async_ld16(&lV[buf][vRow * VP + vChunk], vBase + kt);
    };

    const v8f vzero = {};
    v8f cacc[4];
#pragma unroll
    for (int nt = 0; nt < 4; nt++) cacc[nt] = vzero;
    float mrow[8], lrow[8];
#pragma unroll
    for (int v = 0; v < 8; v++) { mrow[v] = -1e30f; lrow[v] = 0.f; }

    unsigned short* myP = &lp[wid * 640];

    constexpr int NT = LL / 32;   // 64 k-tiles
    issue(0, 0);
    for (int t = 0; t < NT; ++t) {
        const int buf = t & 1;
        const int kt = t * 32;
        if (t + 1 < NT) { issue(kt + 32, buf ^ 1); wait_async2(); }
        else            { wait_async0(); }
        __syncthreads();

        // ---- scores: two 16x16 tiles (k positions kt..kt+31) ----
        v8f s0 = vzero, s1 = vzero;
#pragma unroll
        for (int s = 0; s < 2; s++) {
            FragBF kb0, kb1;
            int b0 = (ln16)      * KP + s * 32 + hf * 8;
            int b1 = (16 + ln16) * KP + s * 32 + hf * 8;
            kb0.q[0] = *(const uint4*)(&lK[buf][b0]);
            kb0.q[1] = *(const uint4*)(&lK[buf][b0 + 16]);
            kb1.q[0] = *(const uint4*)(&lK[buf][b1]);
            kb1.q[1] = *(const uint4*)(&lK[buf][b1 + 16]);
            s0 = wmma_bf16(qf[s], kb0, s0);
            s1 = wmma_bf16(qf[s], kb1, s1);
        }

        // ---- online softmax (row = hf*8+v, 32 cols across the two tiles) ----
#pragma unroll
        for (int v = 0; v < 8; v++) {
            float a = s0[v], c = s1[v];
            float t2 = fmaxf(a, c);
            t2 = fmaxf(t2, __shfl_xor(t2, 1, 16));
            t2 = fmaxf(t2, __shfl_xor(t2, 2, 16));
            t2 = fmaxf(t2, __shfl_xor(t2, 4, 16));
            t2 = fmaxf(t2, __shfl_xor(t2, 8, 16));
            float mn   = fmaxf(mrow[v], t2);
            float corr = __expf(mrow[v] - mn);
            mrow[v] = mn;
            float p0 = __expf(a - mn);
            float p1 = __expf(c - mn);
            float rs = p0 + p1;
            rs += __shfl_xor(rs, 1, 16);
            rs += __shfl_xor(rs, 2, 16);
            rs += __shfl_xor(rs, 4, 16);
            rs += __shfl_xor(rs, 8, 16);
            lrow[v] = lrow[v] * corr + rs;
#pragma unroll
            for (int nt = 0; nt < 4; nt++) cacc[nt][v] *= corr;
            myP[(hf * 8 + v) * 40 + ln16]      = f2bf(p0);
            myP[(hf * 8 + v) * 40 + 16 + ln16] = f2bf(p1);
        }
        asm volatile("s_wait_dscnt 0x0" ::: "memory");  // in-wave LDS visibility

        // ---- P as A-fragment (16x32) ----
        FragBF pf;
        {
            const unsigned short* pp = myP + ln16 * 40 + hf * 8;
            pf.q[0] = *(const uint4*)pp;
            pf.q[1] = *(const uint4*)(pp + 16);
        }
        // ---- ctx += P * V  (V rows = dk index, contiguous along k-pos) ----
#pragma unroll
        for (int nt = 0; nt < 4; nt++) {
            FragBF vb;
            int bv = (nt * 16 + ln16) * VP + hf * 8;
            vb.q[0] = *(const uint4*)(&lV[buf][bv]);
            vb.q[1] = *(const uint4*)(&lV[buf][bv + 16]);
            cacc[nt] = wmma_bf16(pf, vb, cacc[nt]);
        }
        __syncthreads();   // all waves done with buf before async overwrite
    }

    // ---- normalize and store ctx bf16 [B,L,D] ----
#pragma unroll
    for (int nt = 0; nt < 4; nt++) {
        const int col = h * DKH + nt * 16 + ln16;
#pragma unroll
        for (int v = 0; v < 8; v++) {
            const int l = qr0 + hf * 8 + v;
            float val = cacc[nt][v] / lrow[v];
            ctx[((size_t)b * LL + l) * DD + col] = f2bf(val);
        }
    }
}

// ---------------------------------------------------------------------------
// LayerNorm over D=1024: one row per block; writes fp32 (+optional bf16)
// ---------------------------------------------------------------------------
template<bool WB>
__global__ __launch_bounds__(256)
void ln_kernel(const float* __restrict__ in, const float* __restrict__ gamma,
               const float* __restrict__ beta, float* __restrict__ outF,
               unsigned short* __restrict__ outB) {
    __shared__ float red[16];
    const int row = blockIdx.x;
    const float* p = in + (size_t)row * DD;
    float s = 0.f, s2 = 0.f, vals[4];
#pragma unroll
    for (int i = 0; i < 4; i++) {
        float v = p[threadIdx.x + 256 * i];
        vals[i] = v; s += v; s2 += v * v;
    }
#pragma unroll
    for (int o = 16; o > 0; o >>= 1) {
        s  += __shfl_xor(s,  o, 32);
        s2 += __shfl_xor(s2, o, 32);
    }
    const int wid = threadIdx.x >> 5, lane = threadIdx.x & 31;
    if (lane == 0) { red[wid] = s; red[8 + wid] = s2; }
    __syncthreads();
    float ts = 0.f, ts2 = 0.f;
#pragma unroll
    for (int i = 0; i < 8; i++) { ts += red[i]; ts2 += red[8 + i]; }
    const float mean = ts * (1.f / DD);
    const float var  = ts2 * (1.f / DD) - mean * mean;
    const float rstd = rsqrtf(var + 1e-6f);
#pragma unroll
    for (int i = 0; i < 4; i++) {
        const int c = threadIdx.x + 256 * i;
        float o = gamma[c] * (vals[i] - mean) * rstd + beta[c];
        outF[(size_t)row * DD + c] = o;
        if (WB) outB[(size_t)row * DD + c] = f2bf(o);
    }
}

// ---------------------------------------------------------------------------
// Host launcher
// ---------------------------------------------------------------------------
extern "C" void kernel_launch(void* const* d_in, const int* in_sizes, int n_in,
                              void* d_out, int out_size, void* d_ws, size_t ws_size,
                              hipStream_t stream) {
    (void)in_sizes; (void)n_in; (void)out_size; (void)ws_size;
    const float* x   = (const float*)d_in[0];
    const float* w_q = (const float*)d_in[1];
    const float* w_k = (const float*)d_in[2];
    const float* w_v = (const float*)d_in[3];
    const float* w_o = (const float*)d_in[4];
    const float* b_o = (const float*)d_in[5];
    const float* g1  = (const float*)d_in[6];
    const float* be1 = (const float*)d_in[7];
    const float* g2  = (const float*)d_in[8];
    const float* be2 = (const float*)d_in[9];
    const float* w1  = (const float*)d_in[10];
    const float* b1  = (const float*)d_in[11];
    const float* w2  = (const float*)d_in[12];
    const float* b2  = (const float*)d_in[13];
    float* out = (float*)d_out;

    char* ws = (char*)d_ws;
    constexpr size_t MB = 1024 * 1024;
    unsigned short* xb   = (unsigned short*)(ws + 0);
    unsigned short* wqt  = (unsigned short*)(ws + 8  * MB);
    unsigned short* wkt  = (unsigned short*)(ws + 10 * MB);
    unsigned short* wvt  = (unsigned short*)(ws + 12 * MB);
    unsigned short* wot  = (unsigned short*)(ws + 14 * MB);
    unsigned short* w1t  = (unsigned short*)(ws + 16 * MB);
    unsigned short* w2t  = (unsigned short*)(ws + 24 * MB);
    float*          qlin = (float*)(ws + 32 * MB);
    float*          klin = (float*)(ws + 48 * MB);
    float*          vlin = (float*)(ws + 64 * MB);
    unsigned short* qh   = (unsigned short*)(ws + 80  * MB);
    unsigned short* kh   = (unsigned short*)(ws + 88  * MB);
    unsigned short* vt   = (unsigned short*)(ws + 96  * MB);
    unsigned short* ctxb = (unsigned short*)(ws + 104 * MB);
    unsigned short* ffb  = (unsigned short*)(ws + 112 * MB);   // 32 MB
    // safe aliases (producers dead before reuse):
    float*          hpre = vlin;                      // attn-out pre-LN
    float*          hbuf = qlin;                      // h (fp32 residual)
    unsigned short* hb   = (unsigned short*)klin;     // h (bf16 for FFN)
    float*          gpre = vlin;                      // ffn-out pre-LN

    const dim3 blk256(256);
    const dim3 tb(32, 8);

    // 1) precision conversion + weight transposes
    cvtx_kernel<<<MM * DD / 4 / 256, blk256, 0, stream>>>(x, xb);
    wtrans_kernel<<<dim3(32, 32),  tb, 0, stream>>>(w_q, wqt, DD, DD);
    wtrans_kernel<<<dim3(32, 32),  tb, 0, stream>>>(w_k, wkt, DD, DD);
    wtrans_kernel<<<dim3(32, 32),  tb, 0, stream>>>(w_v, wvt, DD, DD);
    wtrans_kernel<<<dim3(32, 32),  tb, 0, stream>>>(w_o, wot, DD, DD);
    wtrans_kernel<<<dim3(32, 128), tb, 0, stream>>>(w1,  w1t, DD, FF);
    wtrans_kernel<<<dim3(128, 32), tb, 0, stream>>>(w2,  w2t, FF, DD);

    // 2) Q/K/V projections (WMMA)
    gemm_bf16<false,false,false,false><<<(MM/128)*(DD/128), blk256, 0, stream>>>(
        xb, wqt, nullptr, nullptr, qlin, nullptr, MM, DD, DD);
    gemm_bf16<false,false,false,false><<<(MM/128)*(DD/128), blk256, 0, stream>>>(
        xb, wkt, nullptr, nullptr, klin, nullptr, MM, DD, DD);
    gemm_bf16<false,false,false,false><<<(MM/128)*(DD/128), blk256, 0, stream>>>(
        xb, wvt, nullptr, nullptr, vlin, nullptr, MM, DD, DD);

    // 3) RoPE + head-major repack; V transpose
    rope_kernel<<<(BB*HH*LL*32) / 256, blk256, 0, stream>>>(qlin, klin, qh, kh);
    vtrans_kernel<<<(BB*HH*DKH*LL) / 256, blk256, 0, stream>>>(vlin, vt);

    // 4) attention (WMMA flash, async K/V staging)
    attn_kernel<<<BB * HH * (LL / 128), blk256, 0, stream>>>(qh, kh, vt, ctxb);

    // 5) output projection + bias + residual(x)
    gemm_bf16<true,true,false,false><<<(MM/128)*(DD/128), blk256, 0, stream>>>(
        ctxb, wot, b_o, x, hpre, nullptr, MM, DD, DD);
    ln_kernel<true><<<MM, blk256, 0, stream>>>(hpre, g1, be1, hbuf, hb);

    // 6) FFN
    gemm_bf16<true,false,true,true><<<(MM/128)*(FF/128), blk256, 0, stream>>>(
        hb, w1t, b1, nullptr, nullptr, ffb, MM, FF, DD);
    gemm_bf16<true,true,false,false><<<(MM/128)*(DD/128), blk256, 0, stream>>>(
        ffb, w2t, b2, hbuf, gpre, nullptr, MM, DD, FF);
    ln_kernel<false><<<MM, blk256, 0, stream>>>(gpre, g2, be2, out, nullptr);
}
